// MusicSelfAttention_74500502716601
// MI455X (gfx1250) — compile-verified, matrix operand order checked by reading, not run on hardware
//
#include <hip/hip_runtime.h>

typedef unsigned short u16;
typedef __bf16 bf16_t;
typedef __attribute__((ext_vector_type(16))) bf16_t v16bf;
typedef __attribute__((ext_vector_type(8)))  float  v8f;
typedef __attribute__((ext_vector_type(4)))  int    v4i;

#define BATCH 2
#define SEQ   2048
#define DIM   2048
#define NH    16
#define HD    128
#define MTOT  (BATCH * SEQ)   // 4096

// ---- CDNA5 async global->LDS copy (ASYNCcnt) ------------------------------
#if defined(__gfx1250__) && __has_builtin(__builtin_amdgcn_global_load_async_to_lds_b128)
#define HAVE_ASYNC_LDS 1
#else
#define HAVE_ASYNC_LDS 0
#endif

#if HAVE_ASYNC_LDS
typedef __attribute__((address_space(1))) v4i* g1v4;   // global (AS1) int4*
typedef __attribute__((address_space(3))) v4i* l3v4;   // LDS (AS3) int4*
static __device__ inline void async_cp16B(const u16* g, u16* l) {
    v4i* gg = (v4i*)g;   // strip const / retype in generic AS
    v4i* ll = (v4i*)l;
    __builtin_amdgcn_global_load_async_to_lds_b128((g1v4)gg, (l3v4)ll, 0, 0);
}
#if __has_builtin(__builtin_amdgcn_s_wait_asynccnt)
#define WAIT_ASYNC(n) __builtin_amdgcn_s_wait_asynccnt(n)
#else
#define WAIT_ASYNC(n) asm volatile("s_wait_asynccnt %0" ::"n"(n) : "memory")
#endif
#endif

static __device__ inline u16 f2bf(float f) {
    bf16_t h = (bf16_t)f;             // RNE via native cvt
    return __builtin_bit_cast(u16, h);
}
static __device__ inline float bf2f(u16 h) {
    return __uint_as_float(((unsigned)h) << 16);
}
static __device__ inline unsigned pack2bf(float a, float b) {
    return (unsigned)f2bf(a) | ((unsigned)f2bf(b) << 16);
}
static __device__ inline v8f vzero8() {
    v8f z;
#pragma unroll
    for (int e = 0; e < 8; ++e) z[e] = 0.f;
    return z;
}
static __device__ inline v8f wmma_bf16(v16bf a, v16bf b, v8f c) {
    return __builtin_amdgcn_wmma_f32_16x16x32_bf16(false, a, false, b, (short)0, c, false, false);
}

// DPP16 all-reduce within each 16-lane half (quad swaps, half mirror, row mirror)
static __device__ inline float dpp_red_max16(float v) {
    int x;
    x = __builtin_amdgcn_update_dpp(0, __builtin_bit_cast(int, v), 0xB1, 0xF, 0xF, true);
    v = fmaxf(v, __builtin_bit_cast(float, x));
    x = __builtin_amdgcn_update_dpp(0, __builtin_bit_cast(int, v), 0x4E, 0xF, 0xF, true);
    v = fmaxf(v, __builtin_bit_cast(float, x));
    x = __builtin_amdgcn_update_dpp(0, __builtin_bit_cast(int, v), 0x141, 0xF, 0xF, true);
    v = fmaxf(v, __builtin_bit_cast(float, x));
    x = __builtin_amdgcn_update_dpp(0, __builtin_bit_cast(int, v), 0x140, 0xF, 0xF, true);
    v = fmaxf(v, __builtin_bit_cast(float, x));
    return v;
}
static __device__ inline float dpp_red_sum16(float v) {
    int x;
    x = __builtin_amdgcn_update_dpp(0, __builtin_bit_cast(int, v), 0xB1, 0xF, 0xF, true);
    v += __builtin_bit_cast(float, x);
    x = __builtin_amdgcn_update_dpp(0, __builtin_bit_cast(int, v), 0x4E, 0xF, 0xF, true);
    v += __builtin_bit_cast(float, x);
    x = __builtin_amdgcn_update_dpp(0, __builtin_bit_cast(int, v), 0x141, 0xF, 0xF, true);
    v += __builtin_bit_cast(float, x);
    x = __builtin_amdgcn_update_dpp(0, __builtin_bit_cast(int, v), 0x140, 0xF, 0xF, true);
    v += __builtin_bit_cast(float, x);
    return v;
}

union FragBF { v16bf v; v4i q[2]; };

// ---------------------------------------------------------------- convert (float4 -> 4x bf16)
__global__ void f2bf4_kernel(const float4* __restrict__ src, unsigned* __restrict__ dst, int n4) {
    int i = blockIdx.x * 256 + threadIdx.x;
    if (i >= n4) return;
    float4 v = src[i];
    dst[2 * i]     = pack2bf(v.x, v.y);
    dst[2 * i + 1] = pack2bf(v.z, v.w);
}

// ---------------------------------------------------------------- GEMM
// C[M,N] = A[M,K](bf16) * W[N,K](bf16)^T + bias ; block tile 128x128, K-tile 64,
// async-to-LDS double-buffered pipeline, 8 waves each 32x64 (2x4 wmma tiles)
#define ASTR 72
template <bool OUTF32>
__global__ __launch_bounds__(256) void gemm_bf16(const u16* __restrict__ A,
                                                 const u16* __restrict__ W,
                                                 const float* __restrict__ bias,
                                                 void* __restrict__ Cout,
                                                 int M, int N, int K) {
#if HAVE_ASYNC_LDS
    __shared__ u16 As[2][128 * ASTR];
    __shared__ u16 Ws[2][128 * ASTR];
#else
    __shared__ u16 As[1][128 * ASTR];
    __shared__ u16 Ws[1][128 * ASTR];
#endif
    const int tid  = threadIdx.x;
    const int lane = tid & 31;
    const int w    = tid >> 5;
    const int r    = lane & 15;
    const int half = (lane >> 4) & 1;
    const int mBase = blockIdx.y * 128;
    const int nBase = blockIdx.x * 128;
    const int wm = (w & 3) * 32;
    const int wn = (w >> 2) * 64;

    v8f acc[2][4];
#pragma unroll
    for (int i = 0; i < 2; ++i)
#pragma unroll
        for (int j = 0; j < 4; ++j) acc[i][j] = vzero8();

    auto compute = [&](int buf) {
#pragma unroll
        for (int kk2 = 0; kk2 < 2; ++kk2) {
            FragBF af[2], bfr[4];
#pragma unroll
            for (int i = 0; i < 2; ++i) {
                const u16* rp = &As[buf][(wm + i * 16 + r) * ASTR + kk2 * 32];
                af[i].q[0] = *(const v4i*)(rp + half * 8);
                af[i].q[1] = *(const v4i*)(rp + 16 + half * 8);
            }
#pragma unroll
            for (int j = 0; j < 4; ++j) {
                const u16* rp = &Ws[buf][(wn + j * 16 + r) * ASTR + kk2 * 32];
                bfr[j].q[0] = *(const v4i*)(rp + half * 8);
                bfr[j].q[1] = *(const v4i*)(rp + 16 + half * 8);
            }
#pragma unroll
            for (int i = 0; i < 2; ++i)
#pragma unroll
                for (int j = 0; j < 4; ++j)
                    acc[i][j] = wmma_bf16(af[i].v, bfr[j].v, acc[i][j]);
        }
    };

#if HAVE_ASYNC_LDS
    auto issue = [&](int buf, int k0) {
#pragma unroll
        for (int p = 0; p < 4; ++p) {
            int idx = p * 256 + tid;
            int row = idx >> 3;
            int c8  = (idx & 7) * 8;
            async_cp16B(&A[(size_t)(mBase + row) * K + k0 + c8], &As[buf][row * ASTR + c8]);
            async_cp16B(&W[(size_t)(nBase + row) * K + k0 + c8], &Ws[buf][row * ASTR + c8]);
        }
    };
    issue(0, 0);
    int cur = 0;
    for (int k0 = 64; k0 < K; k0 += 64) {
        issue(cur ^ 1, k0);     // 8 async ops for the next tile
        WAIT_ASYNC(8);          // in-order completion => current tile resident
        __syncthreads();
        compute(cur);
        __syncthreads();        // all waves done before buffer reuse
        cur ^= 1;
    }
    WAIT_ASYNC(0);
    __syncthreads();
    compute(cur);
#else
    v4i pa[4], pw[4];
    auto loadg = [&](int k0) {
#pragma unroll
        for (int p = 0; p < 4; ++p) {
            int idx = p * 256 + tid;
            int row = idx >> 3;
            int c8  = (idx & 7) * 8;
            pa[p] = *(const v4i*)&A[(size_t)(mBase + row) * K + k0 + c8];
            pw[p] = *(const v4i*)&W[(size_t)(nBase + row) * K + k0 + c8];
        }
    };
    auto stolds = [&]() {
#pragma unroll
        for (int p = 0; p < 4; ++p) {
            int idx = p * 256 + tid;
            int row = idx >> 3;
            int c8  = (idx & 7) * 8;
            *(v4i*)&As[0][row * ASTR + c8] = pa[p];
            *(v4i*)&Ws[0][row * ASTR + c8] = pw[p];
        }
    };
    loadg(0);
    stolds();
    for (int k0 = 64; k0 < K; k0 += 64) {
        __syncthreads();
        loadg(k0);
        compute(0);
        __syncthreads();
        stolds();
    }
    __syncthreads();
    compute(0);
#endif

#pragma unroll
    for (int i = 0; i < 2; ++i)
#pragma unroll
        for (int j = 0; j < 4; ++j)
#pragma unroll
            for (int e = 0; e < 8; ++e) {
                int gm = mBase + wm + i * 16 + e + half * 8;
                int gn = nBase + wn + j * 16 + r;
                float v = acc[i][j][e] + bias[gn];
                if (OUTF32) ((float*)Cout)[(size_t)gm * N + gn] = v;
                else        ((u16*)Cout)[(size_t)gm * N + gn]   = f2bf(v);
            }
}

// ---------------------------------------------------------------- RoPE (in place; Q also scaled by 1/sqrt(HD))
__global__ void rope_kernel(u16* __restrict__ qb, u16* __restrict__ kb,
                            const float* __restrict__ freqs, int n) {
    int idx = blockIdx.x * 256 + threadIdx.x;
    if (idx >= n) return;
    int i  = idx & 63;
    int hh = (idx >> 6) & 15;
    int s  = (idx >> 10) & 2047;
    int bb = idx >> 21;
    const float rs = 0.08838834764831845f;  // 1/sqrt(128), folded into Q
    const float* f = freqs + ((size_t)s * 64 + i) * 4;  // [c, -sin, sin, c]
    float f00 = f[0], f01 = f[1], f10 = f[2], f11 = f[3];
    size_t off = ((size_t)(bb * SEQ + s)) * DIM + hh * HD + i * 2;

    unsigned uq = *(const unsigned*)&qb[off];
    float q0 = bf2f((u16)uq), q1 = bf2f((u16)(uq >> 16));
    *(unsigned*)&qb[off] = pack2bf((f00 * q0 + f01 * q1) * rs, (f10 * q0 + f11 * q1) * rs);

    unsigned uk = *(const unsigned*)&kb[off];
    float k0 = bf2f((u16)uk), k1 = bf2f((u16)(uk >> 16));
    *(unsigned*)&kb[off] = pack2bf(f00 * k0 + f01 * k1, f10 * k0 + f11 * k1);
}

// ---------------------------------------------------------------- V transpose: (B,S,H,HD)->(B,H,HD,S)
#define TS 130
__global__ __launch_bounds__(256) void transpose_v(const u16* __restrict__ vb, u16* __restrict__ vtb) {
    __shared__ u16 tile[32 * TS];
    const int tid  = threadIdx.x;
    const int sblk = blockIdx.x & 63;
    const int h    = (blockIdx.x >> 6) & 15;
    const int b    = blockIdx.x >> 10;
#pragma unroll
    for (int p = 0; p < 8; ++p) {
        int idx = p * 256 + tid;
        int s   = idx >> 6;
        int hu  = idx & 63;
        unsigned u = *(const unsigned*)&vb[((size_t)(b * SEQ + sblk * 32 + s)) * DIM + h * HD + hu * 2];
        *(unsigned*)&tile[s * TS + hu * 2] = u;
    }
    __syncthreads();
#pragma unroll
    for (int p = 0; p < 16; ++p) {
        int idx = p * 256 + tid;
        int hd  = idx >> 5;
        int s   = idx & 31;
        vtb[((size_t)((b * NH + h) * HD + hd)) * SEQ + sblk * 32 + s] = tile[s * TS + hd];
    }
}

// ---------------------------------------------------------------- flash attention
#define KSTR 136
#define VSTR 40
#define PSTR 40
__global__ __launch_bounds__(256) void flash_attn(const u16* __restrict__ qb,
                                                  const u16* __restrict__ kb,
                                                  const u16* __restrict__ vtb,
                                                  u16* __restrict__ ob) {
#if HAVE_ASYNC_LDS
    __shared__ u16 Ks[2][32 * KSTR];
    __shared__ u16 Vs[2][128 * VSTR];
#else
    __shared__ u16 Ks[1][32 * KSTR];
    __shared__ u16 Vs[1][128 * VSTR];
#endif
    __shared__ u16 Ps[8 * 16 * PSTR];   // per-wave P scratch: 16 x 32

    const int tid  = threadIdx.x;
    const int lane = tid & 31;
    const int w    = tid >> 5;
    const int r    = lane & 15;
    const int half = (lane >> 4) & 1;
    const int qblk = blockIdx.x & 15;
    const int h    = (blockIdx.x >> 4) & 15;
    const int b    = blockIdx.x >> 8;

    // persistent Q fragments (pre-scaled by 1/sqrt(HD) in rope_kernel)
    FragBF qf[4];
    {
        const u16* qrow = qb + ((size_t)(b * SEQ + qblk * 128 + w * 16 + r)) * DIM + h * HD;
#pragma unroll
        for (int kk = 0; kk < 4; ++kk) {
            qf[kk].q[0] = *(const v4i*)(qrow + kk * 32 + half * 8);
            qf[kk].q[1] = *(const v4i*)(qrow + kk * 32 + 16 + half * 8);
        }
    }

    float mrow[8], lrow[8];   // lrow: per-lane partial sums (reduced once at end)
    v8f oacc[8];
#pragma unroll
    for (int i = 0; i < 8; ++i) { mrow[i] = -1e30f; lrow[i] = 0.f; }
#pragma unroll
    for (int j = 0; j < 8; ++j) oacc[j] = vzero8();

    auto compute = [&](int buf) {
        // logits S(16x32) = Q(16x128) K_tile^T
        v8f sacc[2];
#pragma unroll
        for (int t2 = 0; t2 < 2; ++t2) {
            sacc[t2] = vzero8();
#pragma unroll
            for (int kk = 0; kk < 4; ++kk) {
                FragBF kf;
                const u16* kp = &Ks[buf][(t2 * 16 + r) * KSTR + kk * 32];
                kf.q[0] = *(const v4i*)(kp + half * 8);
                kf.q[1] = *(const v4i*)(kp + 16 + half * 8);
                sacc[t2] = wmma_bf16(qf[kk].v, kf.v, sacc[t2]);
            }
        }
        // online softmax: DPP row-max, per-lane partial row-sum
#pragma unroll
        for (int i = 0; i < 8; ++i) {
            float s0 = sacc[0][i];
            float s1 = sacc[1][i];
            float mx = dpp_red_max16(fmaxf(s0, s1));
            float mnew  = fmaxf(mrow[i], mx);
            float alpha = __expf(mrow[i] - mnew);
            float p0 = __expf(s0 - mnew);
            float p1 = __expf(s1 - mnew);
            lrow[i] = lrow[i] * alpha + (p0 + p1);
            mrow[i] = mnew;
#pragma unroll
            for (int j = 0; j < 8; ++j) oacc[j][i] *= alpha;
            u16* pwp = &Ps[(w * 16 + i + half * 8) * PSTR];
            pwp[r]      = f2bf(p0);
            pwp[16 + r] = f2bf(p1);
        }
        // O(16x128) += P(16x32) V_tile(32x128)
        FragBF pf;
        {
            const u16* pr = &Ps[(w * 16 + r) * PSTR];
            pf.q[0] = *(const v4i*)(pr + half * 8);
            pf.q[1] = *(const v4i*)(pr + 16 + half * 8);
        }
#pragma unroll
        for (int j = 0; j < 8; ++j) {
            FragBF vf;
            const u16* vp = &Vs[buf][(j * 16 + r) * VSTR];
            vf.q[0] = *(const v4i*)(vp + half * 8);
            vf.q[1] = *(const v4i*)(vp + 16 + half * 8);
            oacc[j] = wmma_bf16(pf.v, vf.v, oacc[j]);
        }
    };

#if HAVE_ASYNC_LDS
    auto issueKV = [&](int buf, int kt) {
#pragma unroll
        for (int p = 0; p < 2; ++p) {
            int idx  = p * 256 + tid;
            int rowK = idx >> 4;
            int cK   = (idx & 15) * 8;
            async_cp16B(&kb[((size_t)(b * SEQ + kt + rowK)) * DIM + h * HD + cK],
                        &Ks[buf][rowK * KSTR + cK]);
            int rowV = idx >> 2;
            int cV   = (idx & 3) * 8;
            async_cp16B(&vtb[((size_t)((b * NH + h) * HD + rowV)) * SEQ + kt + cV],
                        &Vs[buf][rowV * VSTR + cV]);
        }
    };
    issueKV(0, 0);
    int cur = 0;
    for (int kt = 32; kt < SEQ; kt += 32) {
        issueKV(cur ^ 1, kt);   // 4 async ops for the next K/V tile
        WAIT_ASYNC(4);
        __syncthreads();
        compute(cur);
        __syncthreads();
        cur ^= 1;
    }
    WAIT_ASYNC(0);
    __syncthreads();
    compute(cur);
#else
    v4i pk[2], pv[2];
    auto loadKV = [&](int kt) {
#pragma unroll
        for (int p = 0; p < 2; ++p) {
            int idx  = p * 256 + tid;
            int rowK = idx >> 4;
            int cK   = (idx & 15) * 8;
            pk[p] = *(const v4i*)&kb[((size_t)(b * SEQ + kt + rowK)) * DIM + h * HD + cK];
            int rowV = idx >> 2;
            int cV   = (idx & 3) * 8;
            pv[p] = *(const v4i*)&vtb[((size_t)((b * NH + h) * HD + rowV)) * SEQ + kt + cV];
        }
    };
    auto stoKV = [&]() {
#pragma unroll
        for (int p = 0; p < 2; ++p) {
            int idx  = p * 256 + tid;
            int rowK = idx >> 4;
            int cK   = (idx & 15) * 8;
            *(v4i*)&Ks[0][rowK * KSTR + cK] = pk[p];
            int rowV = idx >> 2;
            int cV   = (idx & 3) * 8;
            *(v4i*)&Vs[0][rowV * VSTR + cV] = pv[p];
        }
    };
    loadKV(0);
    stoKV();
    for (int kt = 32; kt < SEQ; kt += 32) {
        __syncthreads();
        loadKV(kt);
        compute(0);
        __syncthreads();
        stoKV();
    }
    __syncthreads();
    compute(0);
#endif

    // reduce per-lane partial sums across 16-lane halves, normalize, store
#pragma unroll
    for (int i = 0; i < 8; ++i) lrow[i] = dpp_red_sum16(lrow[i]);
#pragma unroll
    for (int j = 0; j < 8; ++j)
#pragma unroll
        for (int i = 0; i < 8; ++i) {
            float ov = oacc[j][i] / lrow[i];
            ob[((size_t)(b * SEQ + qblk * 128 + w * 16 + i + half * 8)) * DIM + h * HD + j * 16 + r] =
                f2bf(ov);
        }
}

// ---------------------------------------------------------------- launch
extern "C" void kernel_launch(void* const* d_in, const int* in_sizes, int n_in,
                              void* d_out, int out_size, void* d_ws, size_t ws_size,
                              hipStream_t stream) {
    const float* x     = (const float*)d_in[0];
    const float* freqs = (const float*)d_in[1];
    const float* wq    = (const float*)d_in[2];
    const float* bq    = (const float*)d_in[3];
    const float* wk    = (const float*)d_in[4];
    const float* bk    = (const float*)d_in[5];
    const float* wv    = (const float*)d_in[6];
    const float* bv    = (const float*)d_in[7];
    const float* wo    = (const float*)d_in[8];
    const float* bo    = (const float*)d_in[9];
    float* out = (float*)d_out;

    const size_t nX = (size_t)MTOT * DIM;
    const size_t nW = (size_t)DIM * DIM;

    char* p = (char*)d_ws;
    u16* xb  = (u16*)p; p += nX * 2;
    u16* wqb = (u16*)p; p += nW * 2;
    u16* wkb = (u16*)p; p += nW * 2;
    u16* wvb = (u16*)p; p += nW * 2;
    u16* wob = (u16*)p; p += nW * 2;
    u16* qbb = (u16*)p; p += nX * 2;
    u16* kbb = (u16*)p; p += nX * 2;
    u16* vbb = (u16*)p; p += nX * 2;
    u16* vtb = (u16*)p; p += nX * 2;
    u16* obb = (u16*)p; p += nX * 2;

    f2bf4_kernel<<<(int)(nX / 4 + 255) / 256, 256, 0, stream>>>((const float4*)x,  (unsigned*)xb,  (int)(nX / 4));
    f2bf4_kernel<<<(int)(nW / 4 + 255) / 256, 256, 0, stream>>>((const float4*)wq, (unsigned*)wqb, (int)(nW / 4));
    f2bf4_kernel<<<(int)(nW / 4 + 255) / 256, 256, 0, stream>>>((const float4*)wk, (unsigned*)wkb, (int)(nW / 4));
    f2bf4_kernel<<<(int)(nW / 4 + 255) / 256, 256, 0, stream>>>((const float4*)wv, (unsigned*)wvb, (int)(nW / 4));
    f2bf4_kernel<<<(int)(nW / 4 + 255) / 256, 256, 0, stream>>>((const float4*)wo, (unsigned*)wob, (int)(nW / 4));

    dim3 gg(DIM / 128, MTOT / 128);
    gemm_bf16<false><<<gg, 256, 0, stream>>>(xb, wqb, bq, qbb, MTOT, DIM, DIM);
    gemm_bf16<false><<<gg, 256, 0, stream>>>(xb, wkb, bk, kbb, MTOT, DIM, DIM);
    gemm_bf16<false><<<gg, 256, 0, stream>>>(xb, wvb, bv, vbb, MTOT, DIM, DIM);

    int nrope = BATCH * SEQ * NH * (HD / 2);
    rope_kernel<<<(nrope + 255) / 256, 256, 0, stream>>>(qbb, kbb, freqs, nrope);

    transpose_v<<<BATCH * NH * (SEQ / 32), 256, 0, stream>>>(vbb, vtb);

    flash_attn<<<BATCH * NH * (SEQ / 128), 256, 0, stream>>>(qbb, kbb, vtb, obb);

    gemm_bf16<true><<<gg, 256, 0, stream>>>(obb, wob, bo, out, MTOT, DIM, DIM);
}
